// InteractionHead_13185549599248
// MI455X (gfx1250) — matrix operand order repaired and checked by reading, not compiled, not run on hardware
//
#include <hip/hip_runtime.h>
#include <hip/hip_bf16.h>

// ---------------------------------------------------------------------------
// MI455X (gfx1250) InteractionHead:
//   prior(M,600) * sigmoid(relu(relu(X W1 + b1) W2 + b2) W3 + b3)
// bf16 WMMA GEMMs (v_wmma_f32_16x16x32_bf16, f32 accumulate) with
// double-buffered LDS tiles staged via GLOBAL_LOAD_ASYNC_TO_LDS_B128
// (ASYNCcnt) when the toolchain exposes the builtins.
// ---------------------------------------------------------------------------

typedef __bf16 bf16;
typedef __attribute__((ext_vector_type(16))) __bf16 v16bf;
typedef __attribute__((ext_vector_type(8)))  float  v8f;
typedef __attribute__((ext_vector_type(4)))  int    v4i;

#define AS_GLOBAL __attribute__((address_space(1)))
#define AS_LDS    __attribute__((address_space(3)))

#define BM 128
#define BN 128
#define BK 32
#define LDSK 40   // padded LDS row stride in bf16 (80B): conflict-free frag reads, 16B aligned

#if __has_builtin(__builtin_amdgcn_global_load_async_to_lds_b128) && \
    __has_builtin(__builtin_amdgcn_s_wait_asynccnt)
#define HAS_ASYNC_LDS 1
#else
#define HAS_ASYNC_LDS 0
#endif

// 16-byte tile-chunk copy: async DMA global->LDS when available, else via VGPRs.
__device__ __forceinline__ void cp16_g2l(const bf16* __restrict__ g, bf16* l) {
#if HAS_ASYNC_LDS
    __builtin_amdgcn_global_load_async_to_lds_b128(
        (AS_GLOBAL v4i*)(g),
        (AS_LDS    v4i*)(l), 0, 0);
#else
    *(uint4*)l = *(const uint4*)g;
#endif
}

__device__ __forceinline__ void wait_async_writes() {
#if HAS_ASYNC_LDS
    __builtin_amdgcn_s_wait_asynccnt(0);
#endif
}

// ------------------------- helper kernels ----------------------------------

__global__ void cvt_f32_bf16(const float* __restrict__ in, bf16* __restrict__ out,
                             long long n) {
    long long i = (long long)blockIdx.x * blockDim.x + threadIdx.x;
    const long long stride = (long long)gridDim.x * blockDim.x;
    for (; i < n; i += stride) {
        float v = __builtin_nontemporal_load(in + i);     // streaming pass, keep L2 clean
        __builtin_nontemporal_store((bf16)v, out + i);
    }
}

// W is (Kd x N) row-major f32; write Wt as (Npad x Kd) row-major bf16, zero pad rows n>=N.
__global__ void cvt_wt(const float* __restrict__ W, bf16* __restrict__ Wt,
                       int Kd, int N, int Npad) {
    long long total = (long long)Npad * Kd;
    long long i = (long long)blockIdx.x * blockDim.x + threadIdx.x;
    const long long stride = (long long)gridDim.x * blockDim.x;
    for (; i < total; i += stride) {
        int n = (int)(i / Kd);
        int k = (int)(i % Kd);
        Wt[i] = (n < N) ? (bf16)W[(long long)k * N + n] : (bf16)0.0f;
    }
}

// per-pair score product + gathered object label
__global__ void prior_prep(const float* __restrict__ scores,
                           const int* __restrict__ labels,
                           const int* __restrict__ paired,
                           float* __restrict__ ps, int* __restrict__ lab, int M) {
    int m = blockIdx.x * blockDim.x + threadIdx.x;
    if (m < M) {
        int h = paired[2 * m + 0];
        int o = paired[2 * m + 1];
        ps[m]  = scores[h] * scores[o];
        lab[m] = labels[o];
    }
}

// ------------------------- WMMA GEMM ---------------------------------------
// A : (M x Kd) bf16 row-major, Bt : (Ncols x Kd) bf16 row-major (= B transposed)
// mode 0: outBf[row*ldout+col] = bf16(relu(acc + bias[col]))
// mode 1: col<600 -> outF[row*600+col] = ps[row]*class_map[lab[row]*600+col] * sigmoid(acc+bias[col])
__global__ __launch_bounds__(256) void wmma_gemm_bf16(
    const bf16* __restrict__ A, const bf16* __restrict__ Bt,
    const float* __restrict__ bias,
    bf16* __restrict__ outBf, float* __restrict__ outF,
    const float* __restrict__ ps, const int* __restrict__ lab,
    const float* __restrict__ class_map,
    int Kd, int ldout, int mode)
{
    __shared__ __align__(16) bf16 sA[2][BM * LDSK];
    __shared__ __align__(16) bf16 sB[2][BN * LDSK];

    const int tid   = threadIdx.x;
    const int lane  = tid & 31;
    const int wave  = tid >> 5;     // 0..7
    const int waveM = wave & 3;     // 4 slabs of 32 rows
    const int waveN = wave >> 2;    // 2 slabs of 64 cols
    const int l16   = lane & 15;
    const int half  = lane >> 4;

    // global<->LDS staging map: 512 chunks of 8 bf16; thread t owns chunks t, t+256
    const int rowA0 = tid >> 2;              // 0..63
    const int rowA1 = rowA0 + 64;            // 64..127
    const int colA0 = (tid & 3) * 8;         // 0,8,16,24

    const bf16* __restrict__ gA0 = A  + (long long)blockIdx.x * BM * Kd + (long long)rowA0 * Kd + colA0;
    const bf16* __restrict__ gA1 = A  + (long long)blockIdx.x * BM * Kd + (long long)rowA1 * Kd + colA0;
    const bf16* __restrict__ gB0 = Bt + (long long)blockIdx.y * BN * Kd + (long long)rowA0 * Kd + colA0;
    const bf16* __restrict__ gB1 = Bt + (long long)blockIdx.y * BN * Kd + (long long)rowA1 * Kd + colA0;
    bf16* const lA0 = &sA[0][rowA0 * LDSK + colA0];
    bf16* const lA1 = &sA[0][rowA1 * LDSK + colA0];
    bf16* const lB0 = &sB[0][rowA0 * LDSK + colA0];
    bf16* const lB1 = &sB[0][rowA1 * LDSK + colA0];
    const int bufStride = BM * LDSK;         // elements between buffer 0 and 1

    v8f acc[2][4];
#pragma unroll
    for (int mi = 0; mi < 2; ++mi)
#pragma unroll
        for (int ni = 0; ni < 4; ++ni)
#pragma unroll
            for (int v = 0; v < 8; ++v) acc[mi][ni][v] = 0.0f;

    // stage tile 0 into buffer 0
    cp16_g2l(gA0, lA0);
    cp16_g2l(gA1, lA1);
    cp16_g2l(gB0, lB0);
    cp16_g2l(gB1, lB1);

    int cur = 0;
    for (int k0 = 0; k0 < Kd; k0 += BK) {
        wait_async_writes();     // this wave's in-flight tile writes are done
        __syncthreads();         // all waves: writes visible, prior reads of other buffer done

        // prefetch next tile into the alternate buffer (overlaps with the 8 WMMAs below)
        if (k0 + BK < Kd) {
            const int nb = (cur ^ 1) * bufStride;
            cp16_g2l(gA0 + k0 + BK, lA0 + nb);
            cp16_g2l(gA1 + k0 + BK, lA1 + nb);
            cp16_g2l(gB0 + k0 + BK, lB0 + nb);
            cp16_g2l(gB1 + k0 + BK, lB1 + nb);
        }

        const bf16* __restrict__ tA = sA[cur];
        const bf16* __restrict__ tB = sB[cur];

        // A fragments: lane<16 -> K0..15, lane>=16 -> K16..31
        v16bf af[2];
#pragma unroll
        for (int mi = 0; mi < 2; ++mi) {
            const int r = waveM * 32 + mi * 16 + l16;
            const bf16* p = tA + r * LDSK + half * 16;
            union { v16bf v; uint4 q[2]; } u;
            u.q[0] = *(const uint4*)(p);
            u.q[1] = *(const uint4*)(p + 8);
            af[mi] = u.v;
        }
        // B fragments (Bt rows are output columns): lane<16 -> K0..15, lane>=16 -> K16..31
        v16bf bfr[4];
#pragma unroll
        for (int ni = 0; ni < 4; ++ni) {
            const int c = waveN * 64 + ni * 16 + l16;
            const bf16* p = tB + c * LDSK + half * 16;
            union { v16bf v; uint4 q[2]; } u;
            u.q[0] = *(const uint4*)(p);
            u.q[1] = *(const uint4*)(p + 8);
            bfr[ni] = u.v;
        }
#pragma unroll
        for (int mi = 0; mi < 2; ++mi)
#pragma unroll
            for (int ni = 0; ni < 4; ++ni)
                acc[mi][ni] = __builtin_amdgcn_wmma_f32_16x16x32_bf16(
                    false, af[mi], false, bfr[ni], (short)0, acc[mi][ni], false, false);

        cur ^= 1;
    }

    // epilogue: C/D layout -> row = v + 8*half, col = l16 within each 16x16 tile
#pragma unroll
    for (int mi = 0; mi < 2; ++mi) {
#pragma unroll
        for (int v = 0; v < 8; ++v) {
            const int row = blockIdx.x * BM + waveM * 32 + mi * 16 + half * 8 + v;
#pragma unroll
            for (int ni = 0; ni < 4; ++ni) {
                const int col = blockIdx.y * BN + waveN * 64 + ni * 16 + l16;
                float x = acc[mi][ni][v];
                if (mode == 0) {
                    x += bias[col];
                    x = fmaxf(x, 0.0f);
                    outBf[(long long)row * ldout + col] = (bf16)x;
                } else if (col < 600) {
                    const float logit = x + bias[col];
                    const float pr = ps[row] * class_map[lab[row] * 600 + col];
                    outF[(long long)row * 600 + col] = pr / (1.0f + expf(-logit));
                }
            }
        }
    }
}

// ------------------------- launcher ----------------------------------------

extern "C" void kernel_launch(void* const* d_in, const int* in_sizes, int n_in,
                              void* d_out, int out_size, void* d_ws, size_t ws_size,
                              hipStream_t stream)
{
    const float* pair_features = (const float*)d_in[0];
    const float* scores        = (const float*)d_in[1];
    const int*   labels        = (const int*)d_in[2];
    const int*   paired        = (const int*)d_in[3];
    const float* class_map     = (const float*)d_in[4];
    const float* W1            = (const float*)d_in[5];
    const float* b1            = (const float*)d_in[6];
    const float* W2            = (const float*)d_in[7];
    const float* b2            = (const float*)d_in[8];
    const float* W3            = (const float*)d_in[9];
    const float* b3            = (const float*)d_in[10];
    (void)in_sizes; (void)n_in; (void)out_size; (void)ws_size;

    const int M = 4096, FIN = 12544, REP = 1024, K = 600, KP = 640;

    char* ws = (char*)d_ws;
    size_t off = 0;
    auto take = [&](size_t bytes) -> char* {
        char* p = ws + off;
        off = (off + bytes + 255) & ~(size_t)255;
        return p;
    };
    bf16*  Xb  = (bf16*)take((size_t)M  * FIN * 2);   // X in bf16
    bf16*  W1t = (bf16*)take((size_t)REP * FIN * 2);  // W1^T
    bf16*  W2t = (bf16*)take((size_t)REP * REP * 2);  // W2^T
    bf16*  W3t = (bf16*)take((size_t)KP  * REP * 2);  // W3^T zero-padded to 640 rows
    bf16*  H1  = (bf16*)take((size_t)M  * REP * 2);
    bf16*  H2  = (bf16*)take((size_t)M  * REP * 2);
    float* ps  = (float*)take((size_t)M * sizeof(float));
    int*   lab = (int*)  take((size_t)M * sizeof(int));

    cvt_f32_bf16<<<4096, 256, 0, stream>>>(pair_features, Xb, (long long)M * FIN);
    cvt_wt<<<2048, 256, 0, stream>>>(W1, W1t, FIN, REP, REP);
    cvt_wt<<<512,  256, 0, stream>>>(W2, W2t, REP, REP, REP);
    cvt_wt<<<512,  256, 0, stream>>>(W3, W3t, REP, K,  KP);
    prior_prep<<<M / 256, 256, 0, stream>>>(scores, labels, paired, ps, lab, M);

    // GEMM1: (4096x12544)x(12544x1024) + b1, ReLU -> H1 (bf16)
    wmma_gemm_bf16<<<dim3(M / BM, REP / BN), dim3(256), 0, stream>>>(
        Xb, W1t, b1, H1, nullptr, nullptr, nullptr, nullptr, FIN, REP, 0);
    // GEMM2: (4096x1024)x(1024x1024) + b2, ReLU -> H2 (bf16)
    wmma_gemm_bf16<<<dim3(M / BM, REP / BN), dim3(256), 0, stream>>>(
        H1, W2t, b2, H2, nullptr, nullptr, nullptr, nullptr, REP, REP, 0);
    // GEMM3: (4096x1024)x(1024x640pad) + b3, fused prior*sigmoid -> d_out (f32, stride 600)
    wmma_gemm_bf16<<<dim3(M / BM, KP / BN), dim3(256), 0, stream>>>(
        H2, W3t, b3, nullptr, (float*)d_out, ps, lab, class_map, REP, K, 1);
}
